// TransformerBlock_91293824844144
// MI455X (gfx1250) — compile-verified, matrix operand order checked by reading, not run on hardware
//
#include <hip/hip_runtime.h>

// ---------------------------------------------------------------------------
// Types for WMMA bf16 16x16x32 and TDM descriptors (gfx1250 / CDNA5, wave32)
// ---------------------------------------------------------------------------
typedef __bf16          v16bf __attribute__((ext_vector_type(16)));
typedef float           v8f   __attribute__((ext_vector_type(8)));
typedef unsigned short  u16x8 __attribute__((ext_vector_type(8)));
typedef unsigned int    u32x4 __attribute__((ext_vector_type(4)));
typedef int             i32x4 __attribute__((ext_vector_type(4)));
typedef int             i32x8 __attribute__((ext_vector_type(8)));

union FragAB {
    u16x8 h[2];
    v16bf v;
};

__device__ __forceinline__ unsigned short f32_to_bf16(float f) {
    unsigned int u = __float_as_uint(f);
    u = (u + 0x7FFFu + ((u >> 16) & 1u)) >> 16;
    return (unsigned short)u;
}

__device__ __forceinline__ FragAB load_frag(const unsigned short* __restrict__ row,
                                            int kbase) {
    FragAB f;
    f.h[0] = *(const u16x8*)(row + kbase);
    f.h[1] = *(const u16x8*)(row + kbase + 16);
    return f;
}

__device__ __forceinline__ v8f wmma_bf16(const FragAB& a, const FragAB& b, v8f c) {
    return __builtin_amdgcn_wmma_f32_16x16x32_bf16(
        false, a.v, false, b.v, (short)0, c, false, false);
}

// ---------------------------------------------------------------------------
// Tensor Data Mover: stage a 2-D bf16 tile (rows x cols) from global into LDS.
// Builds the D# (groups 0/1; 2-D so groups 2/3 are zero) and issues
// tensor_load_to_lds. Tracked with TENSORcnt.
// ---------------------------------------------------------------------------
__device__ __forceinline__ void tdm_load_tile_bf16(unsigned lds_addr,
                                                   const void* gaddr,
                                                   unsigned rows,     // tile_dim1
                                                   unsigned cols,     // tile_dim0 (elements)
                                                   unsigned tens_d0,  // tensor dim0 (elements)
                                                   unsigned tens_d1,  // tensor dim1 (rows)
                                                   unsigned stride) { // row stride (elements)
    unsigned long long ga = (unsigned long long)gaddr;
    u32x4 g0;
    g0[0] = 1u;                                   // count=1, user descriptor
    g0[1] = lds_addr;                             // LDS byte address
    g0[2] = (unsigned)(ga & 0xffffffffu);         // global_addr[31:0]
    g0[3] = (unsigned)((ga >> 32) & 0x01ffffffu)  // global_addr[56:32]
            | 0x80000000u;                        // type=2 ("image")
    i32x8 g1;
    g1[0] = (int)0x00010000u;                     // data_size=1 -> 2 bytes/elem
    g1[1] = (int)((tens_d0 & 0xffffu) << 16);     // tensor_dim0[15:0]
    g1[2] = (int)((tens_d0 >> 16) | ((tens_d1 & 0xffffu) << 16)); // d0 hi | d1 lo
    g1[3] = (int)((tens_d1 >> 16) | (cols << 16));// d1 hi | tile_dim0
    g1[4] = (int)rows;                            // tile_dim1 (tile_dim2=0)
    g1[5] = (int)stride;                          // tensor_dim0_stride[31:0]
    g1[6] = 0;                                    // stride hi | dim1_stride lo
    g1[7] = 0;
    i32x4 z4 = {0, 0, 0, 0};
#if __clang_major__ >= 23
    i32x8 z8 = {0, 0, 0, 0, 0, 0, 0, 0};
    __builtin_amdgcn_tensor_load_to_lds(g0, g1, z4, z4, z8, 0);
#else
    __builtin_amdgcn_tensor_load_to_lds(g0, g1, z4, z4, 0);
#endif
}

// ---------------------------------------------------------------------------
// Model dimensions
// ---------------------------------------------------------------------------
#define SEQ     1024
#define BATCH   2
#define TOK     (SEQ * BATCH)      // 2048 tokens
#define HID     2048
#define NQH     32
#define NKVH    8
#define HD      64
#define KVW     (NKVH * HD)        // 512
#define FFN     8192
#define KSTEP   64                 // GEMM k-tile staged per TDM transfer

// ---------------------------------------------------------------------------
// Weight convert + transpose: wt[n*K + k] = bf16(w[k*N + n])
// ---------------------------------------------------------------------------
__global__ void k_cvt_t(const float* __restrict__ w, unsigned short* __restrict__ wt,
                        int K, int N) {
    long long idx = (long long)blockIdx.x * blockDim.x + threadIdx.x;
    if (idx >= (long long)K * N) return;
    int k = (int)(idx % K);
    int n = (int)(idx / K);
    wt[idx] = f32_to_bf16(w[(size_t)k * N + n]);
}

// ---------------------------------------------------------------------------
// RMSNorm (optional residual add)
// ---------------------------------------------------------------------------
__global__ void k_rmsnorm(const float* __restrict__ x, const float* __restrict__ add,
                          const float* __restrict__ w, unsigned short* __restrict__ out) {
    __shared__ float red[256];
    int tok = blockIdx.x;
    const float* xr = x + (size_t)tok * HID;
    const float* ar = add ? add + (size_t)tok * HID : nullptr;
    float vals[8];
    float ss = 0.f;
    for (int j = 0; j < 8; j++) {
        int c = threadIdx.x + j * 256;
        float v = xr[c] + (ar ? ar[c] : 0.f);
        vals[j] = v;
        ss += v * v;
    }
    red[threadIdx.x] = ss;
    __syncthreads();
    for (int st = 128; st > 0; st >>= 1) {
        if (threadIdx.x < st) red[threadIdx.x] += red[threadIdx.x + st];
        __syncthreads();
    }
    float rinv = rsqrtf(red[0] / (float)HID + 1e-5f);
    for (int j = 0; j < 8; j++) {
        int c = threadIdx.x + j * 256;
        out[(size_t)tok * HID + c] = f32_to_bf16(vals[j] * rinv * w[c]);
    }
}

// ---------------------------------------------------------------------------
// WMMA GEMM with TDM-staged, double-buffered LDS tiles.
// C[M,N] (f32, + optional residual) = A[M,K] bf16 * B^T[N,K] bf16
// Block: 256 threads = 8 waves; block tile 32(M) x 128(N); K staged in
// 64-wide slices: Ash 32x64, Bsh 128x64 per buffer (40 KB total LDS).
// Wave 0 drives the Tensor Data Mover (2 descriptors per stage) and
// synchronizes with s_wait_tensorcnt; all waves consume via ds_load_b128.
// grid = (N/128, M/32)
// ---------------------------------------------------------------------------
__global__ void k_gemm_bf16(const unsigned short* __restrict__ A,
                            const unsigned short* __restrict__ Bt,
                            float* __restrict__ C,
                            const float* __restrict__ resid,
                            int M, int N, int K) {
    __shared__ __align__(16) unsigned short Ash[2][32][KSTEP];
    __shared__ __align__(16) unsigned short Bsh[2][128][KSTEP];

    int tid  = threadIdx.x;
    int lane = tid & 31;
    int w    = tid >> 5;
    int l16  = lane & 15;
    int h2   = lane >> 4;
    int kb   = h2 << 3;

    int rowBlk = blockIdx.y * 32;
    int colBlk = blockIdx.x * 128;
    int tmr = (w >> 2) * 16;     // wave's A-row offset within block tile
    int tnr = (w & 3) * 32;      // wave's N-col offset within block tile

    const unsigned short* Ab = A  + (size_t)rowBlk * K;
    const unsigned short* Bb = Bt + (size_t)colBlk * K;

    v8f acc0, acc1;
    for (int r = 0; r < 8; r++) { acc0[r] = 0.f; acc1[r] = 0.f; }

    int nk = K / KSTEP;

    if (tid < 32) {  // wave 0 drives the TDM
        tdm_load_tile_bf16((unsigned)(unsigned long long)&Ash[0][0][0],
                           Ab, 32, KSTEP, (unsigned)K, 32u, (unsigned)K);
        tdm_load_tile_bf16((unsigned)(unsigned long long)&Bsh[0][0][0],
                           Bb, 128, KSTEP, (unsigned)K, 128u, (unsigned)K);
    }

    for (int i = 0; i < nk; i++) {
        int buf = i & 1;
        if (tid < 32) {
            if (i + 1 < nk) {
                const unsigned short* An = Ab + (size_t)(i + 1) * KSTEP;
                const unsigned short* Bn = Bb + (size_t)(i + 1) * KSTEP;
                tdm_load_tile_bf16((unsigned)(unsigned long long)&Ash[buf ^ 1][0][0],
                                   An, 32, KSTEP, (unsigned)K, 32u, (unsigned)K);
                tdm_load_tile_bf16((unsigned)(unsigned long long)&Bsh[buf ^ 1][0][0],
                                   Bn, 128, KSTEP, (unsigned)K, 128u, (unsigned)K);
                __builtin_amdgcn_s_wait_tensorcnt(2);  // current stage landed
            } else {
                __builtin_amdgcn_s_wait_tensorcnt(0);
            }
        }
        __syncthreads();   // staged tiles visible to all waves

#pragma unroll
        for (int kk = 0; kk < KSTEP; kk += 32) {
            FragAB a, b0, b1;
            a.h[0]  = *(const u16x8*)&Ash[buf][tmr + l16][kk + kb];
            a.h[1]  = *(const u16x8*)&Ash[buf][tmr + l16][kk + kb + 16];
            b0.h[0] = *(const u16x8*)&Bsh[buf][tnr + l16][kk + kb];
            b0.h[1] = *(const u16x8*)&Bsh[buf][tnr + l16][kk + kb + 16];
            b1.h[0] = *(const u16x8*)&Bsh[buf][tnr + 16 + l16][kk + kb];
            b1.h[1] = *(const u16x8*)&Bsh[buf][tnr + 16 + l16][kk + kb + 16];
            acc0 = wmma_bf16(a, b0, acc0);
            acc1 = wmma_bf16(a, b1, acc1);
        }
        __syncthreads();   // all reads of buf done before it is re-staged
    }

    int coln0 = colBlk + tnr + l16;
    int coln1 = coln0 + 16;
    for (int r = 0; r < 8; r++) {
        int m = rowBlk + tmr + r + 8 * h2;
        size_t i0 = (size_t)m * N + coln0;
        size_t i1 = (size_t)m * N + coln1;
        C[i0] = acc0[r] + (resid ? resid[i0] : 0.f);
        C[i1] = acc1[r] + (resid ? resid[i1] : 0.f);
    }
}

// ---------------------------------------------------------------------------
// RoPE (base 500000, head_dim 64) fp32 -> bf16
// ---------------------------------------------------------------------------
__global__ void k_rope(const float* __restrict__ x, unsigned short* __restrict__ out,
                       int nheads) {
    long long idx = (long long)blockIdx.x * blockDim.x + threadIdx.x;
    int i   = (int)(idx & 31);
    int hh  = (int)((idx >> 5) % nheads);
    int tok = (int)(idx / (32LL * nheads));
    if (tok >= TOK) return;
    int pos = tok & (SEQ - 1);
    float inv = powf(500000.0f, -((float)(2 * i) / 64.0f));
    float ang = (float)pos * inv;
    float c = cosf(ang), s = sinf(ang);
    size_t base = (size_t)tok * (nheads * HD) + (size_t)hh * HD;
    float xa = x[base + i];
    float xb = x[base + i + 32];
    out[base + i]      = f32_to_bf16(xa * c - xb * s);
    out[base + i + 32] = f32_to_bf16(xb * c + xa * s);
}

// ---------------------------------------------------------------------------
// Transpose V per kv-head: vt[b][h][d][s] = bf16(v[(b*SEQ+s)*KVW + h*HD + d])
// ---------------------------------------------------------------------------
__global__ void k_vt(const float* __restrict__ v, unsigned short* __restrict__ vt) {
    int idx = blockIdx.x * blockDim.x + threadIdx.x;
    if (idx >= BATCH * NKVH * HD * SEQ) return;
    int s  = idx & (SEQ - 1);
    int d  = (idx >> 10) & 63;
    int hh = (idx >> 16) & 7;
    int b  = idx >> 19;
    vt[idx] = f32_to_bf16(v[(size_t)(b * SEQ + s) * KVW + hh * HD + d]);
}

// ---------------------------------------------------------------------------
// Flash-style causal GQA attention (1 wave per 16-row q-tile).
// Q*K^T and P*V on WMMA; online softmax with shfl row-reductions;
// P re-laid out C-tile -> A-fragment through LDS.
// ---------------------------------------------------------------------------
__global__ void k_attn(const unsigned short* __restrict__ Qb,   // [TOK, HID]
                       const unsigned short* __restrict__ Kb,   // [TOK, KVW]
                       const unsigned short* __restrict__ Vt,   // [B][NKVH][HD][SEQ]
                       unsigned short* __restrict__ Ctx) {      // [TOK, HID]
    __shared__ __align__(16) unsigned short Ps[16 * 32];

    int lane = threadIdx.x & 31;
    int l16  = lane & 15;
    int h2   = lane >> 4;
    int kb   = h2 << 3;

    int bh  = blockIdx.x;
    int b   = bh >> 5;
    int h   = bh & 31;
    int kvh = h >> 2;
    int qt  = blockIdx.y;
    int q0  = qt * 16;

    const unsigned short* qrow = Qb + (size_t)(b * SEQ + q0 + l16) * HID + h * HD;
    FragAB qf0 = load_frag(qrow, 0 + kb);
    FragAB qf1 = load_frag(qrow, 32 + kb);

    float m[8], l[8];
    v8f cacc[4];
    for (int r = 0; r < 8; r++) { m[r] = -3.0e38f; l[r] = 0.f; }
    for (int d = 0; d < 4; d++)
        for (int r = 0; r < 8; r++) cacc[d][r] = 0.f;

    int nk = (q0 + 47) >> 5;
    for (int kt = 0; kt < nk; kt++) {
        v8f s0, s1;
        for (int r = 0; r < 8; r++) { s0[r] = 0.f; s1[r] = 0.f; }
        const unsigned short* krow0 =
            Kb + (size_t)(b * SEQ + kt * 32 + l16) * KVW + kvh * HD;
        const unsigned short* krow1 = krow0 + (size_t)16 * KVW;
        {
            FragAB kf0 = load_frag(krow0, 0 + kb);
            FragAB kf1 = load_frag(krow1, 0 + kb);
            s0 = wmma_bf16(qf0, kf0, s0);
            s1 = wmma_bf16(qf0, kf1, s1);
            kf0 = load_frag(krow0, 32 + kb);
            kf1 = load_frag(krow1, 32 + kb);
            s0 = wmma_bf16(qf1, kf0, s0);
            s1 = wmma_bf16(qf1, kf1, s1);
        }

        for (int r = 0; r < 8; r++) {
            int rowq = q0 + r + 8 * h2;
            int c0 = kt * 32 + l16;
            int c1 = c0 + 16;
            float v0 = (c0 <= rowq) ? s0[r] * 0.125f : -1.25e8f;
            float v1 = (c1 <= rowq) ? s1[r] * 0.125f : -1.25e8f;
            float mx = fmaxf(v0, v1);
            mx = fmaxf(mx, __shfl_xor(mx, 1, 32));
            mx = fmaxf(mx, __shfl_xor(mx, 2, 32));
            mx = fmaxf(mx, __shfl_xor(mx, 4, 32));
            mx = fmaxf(mx, __shfl_xor(mx, 8, 32));
            float mn = fmaxf(m[r], mx);
            float p0 = __expf(v0 - mn);
            float p1 = __expf(v1 - mn);
            float rs = p0 + p1;
            rs += __shfl_xor(rs, 1, 32);
            rs += __shfl_xor(rs, 2, 32);
            rs += __shfl_xor(rs, 4, 32);
            rs += __shfl_xor(rs, 8, 32);
            float fac = __expf(m[r] - mn);
            l[r] = l[r] * fac + rs;
            m[r] = mn;
            for (int d = 0; d < 4; d++) cacc[d][r] *= fac;
            Ps[(r + 8 * h2) * 32 + l16]      = f32_to_bf16(p0);
            Ps[(r + 8 * h2) * 32 + 16 + l16] = f32_to_bf16(p1);
        }
        __syncthreads();

        FragAB pf;
        pf.h[0] = *(const u16x8*)(Ps + l16 * 32 + kb);
        pf.h[1] = *(const u16x8*)(Ps + l16 * 32 + kb + 16);

        const unsigned short* vbase =
            Vt + (size_t)((b * NKVH + kvh) * HD) * SEQ + kt * 32;
        for (int d = 0; d < 4; d++) {
            const unsigned short* vrow = vbase + (size_t)(d * 16 + l16) * SEQ;
            FragAB vf = load_frag(vrow, kb);
            cacc[d] = wmma_bf16(pf, vf, cacc[d]);
        }
        __syncthreads();
    }

    for (int d = 0; d < 4; d++) {
        for (int r = 0; r < 8; r++) {
            int tok = b * SEQ + q0 + r + 8 * h2;
            Ctx[(size_t)tok * HID + h * HD + d * 16 + l16] =
                f32_to_bf16(cacc[d][r] / l[r]);
        }
    }
}

// ---------------------------------------------------------------------------
// SiLU(gate) * up -> bf16
// ---------------------------------------------------------------------------
__global__ void k_silu_mul(const float* __restrict__ g, const float* __restrict__ u,
                           unsigned short* __restrict__ out, long long n) {
    long long i = (long long)blockIdx.x * blockDim.x + threadIdx.x;
    if (i >= n) return;
    float gv = g[i];
    float sv = gv / (1.f + __expf(-gv));
    out[i] = f32_to_bf16(sv * u[i]);
}

// ---------------------------------------------------------------------------
// Host-side orchestration
// ---------------------------------------------------------------------------
extern "C" void kernel_launch(void* const* d_in, const int* in_sizes, int n_in,
                              void* d_out, int out_size, void* d_ws, size_t ws_size,
                              hipStream_t stream) {
    const float* hidden  = (const float*)d_in[0];
    const float* wq      = (const float*)d_in[2];
    const float* wk      = (const float*)d_in[3];
    const float* wv      = (const float*)d_in[4];
    const float* wo      = (const float*)d_in[5];
    const float* norm1_w = (const float*)d_in[6];
    const float* norm2_w = (const float*)d_in[7];
    const float* w_gate  = (const float*)d_in[8];
    const float* w_up    = (const float*)d_in[9];
    const float* w_down  = (const float*)d_in[10];
    float* out = (float*)d_out;

    char* base = (char*)d_ws;
    size_t off = 0;
    auto alloc = [&](size_t bytes) {
        size_t r = off;
        off += (bytes + 255) & ~(size_t)255;
        return r;
    };
    unsigned short* wqT = (unsigned short*)(base + alloc((size_t)HID * HID * 2));
    unsigned short* wkT = (unsigned short*)(base + alloc((size_t)HID * KVW * 2));
    unsigned short* wvT = (unsigned short*)(base + alloc((size_t)HID * KVW * 2));
    unsigned short* woT = (unsigned short*)(base + alloc((size_t)HID * HID * 2));
    unsigned short* wgT = (unsigned short*)(base + alloc((size_t)HID * FFN * 2));
    unsigned short* wuT = (unsigned short*)(base + alloc((size_t)HID * FFN * 2));
    unsigned short* wdT = (unsigned short*)(base + alloc((size_t)FFN * HID * 2));
    unsigned short* n1  = (unsigned short*)(base + alloc((size_t)TOK * HID * 2));
    float*          qf  = (float*)(base + alloc((size_t)TOK * HID * 4));
    float*          kf  = (float*)(base + alloc((size_t)TOK * KVW * 4));
    float*          vf  = (float*)(base + alloc((size_t)TOK * KVW * 4));
    unsigned short* qb  = (unsigned short*)(base + alloc((size_t)TOK * HID * 2));
    unsigned short* kbuf= (unsigned short*)(base + alloc((size_t)TOK * KVW * 2));
    unsigned short* vt  = (unsigned short*)(base + alloc((size_t)TOK * KVW * 2));
    unsigned short* ctx = (unsigned short*)(base + alloc((size_t)TOK * HID * 2));
    float*          att = (float*)(base + alloc((size_t)TOK * HID * 4));
    unsigned short* n2  = (unsigned short*)(base + alloc((size_t)TOK * HID * 2));
    float*          gbf = (float*)(base + alloc((size_t)TOK * FFN * 4));
    float*          ubf = (float*)(base + alloc((size_t)TOK * FFN * 4));
    unsigned short* act = (unsigned short*)(base + alloc((size_t)TOK * FFN * 2));

    auto cdiv = [](long long a, long long b) { return (int)((a + b - 1) / b); };

    // 1) weights -> bf16, transposed to [N,K]
    k_cvt_t<<<cdiv((long long)HID * HID, 256), 256, 0, stream>>>(wq, wqT, HID, HID);
    k_cvt_t<<<cdiv((long long)HID * KVW, 256), 256, 0, stream>>>(wk, wkT, HID, KVW);
    k_cvt_t<<<cdiv((long long)HID * KVW, 256), 256, 0, stream>>>(wv, wvT, HID, KVW);
    k_cvt_t<<<cdiv((long long)HID * HID, 256), 256, 0, stream>>>(wo, woT, HID, HID);
    k_cvt_t<<<cdiv((long long)HID * FFN, 256), 256, 0, stream>>>(w_gate, wgT, HID, FFN);
    k_cvt_t<<<cdiv((long long)HID * FFN, 256), 256, 0, stream>>>(w_up, wuT, HID, FFN);
    k_cvt_t<<<cdiv((long long)FFN * HID, 256), 256, 0, stream>>>(w_down, wdT, FFN, HID);

    // 2) rmsnorm1
    k_rmsnorm<<<TOK, 256, 0, stream>>>(hidden, nullptr, norm1_w, n1);

    // 3) Q/K/V projections (TDM + WMMA)
    k_gemm_bf16<<<dim3(HID / 128, TOK / 32), 256, 0, stream>>>(n1, wqT, qf, nullptr, TOK, HID, HID);
    k_gemm_bf16<<<dim3(KVW / 128, TOK / 32), 256, 0, stream>>>(n1, wkT, kf, nullptr, TOK, KVW, HID);
    k_gemm_bf16<<<dim3(KVW / 128, TOK / 32), 256, 0, stream>>>(n1, wvT, vf, nullptr, TOK, KVW, HID);

    // 4) RoPE -> bf16 caches; transpose V per kv-head
    k_rope<<<cdiv((long long)TOK * NQH * 32, 256), 256, 0, stream>>>(qf, qb, NQH);
    k_rope<<<cdiv((long long)TOK * NKVH * 32, 256), 256, 0, stream>>>(kf, kbuf, NKVH);
    k_vt<<<cdiv((long long)BATCH * NKVH * HD * SEQ, 256), 256, 0, stream>>>(vf, vt);

    // 5) attention (WMMA flash)
    k_attn<<<dim3(BATCH * NQH, SEQ / 16), 32, 0, stream>>>(qb, kbuf, vt, ctx);

    // 6) output projection
    k_gemm_bf16<<<dim3(HID / 128, TOK / 32), 256, 0, stream>>>(ctx, woT, att, nullptr, TOK, HID, HID);

    // 7) add + rmsnorm2
    k_rmsnorm<<<TOK, 256, 0, stream>>>(att, hidden, norm2_w, n2);

    // 8) FFN gate/up (TDM + WMMA)
    k_gemm_bf16<<<dim3(FFN / 128, TOK / 32), 256, 0, stream>>>(n2, wgT, gbf, nullptr, TOK, FFN, HID);
    k_gemm_bf16<<<dim3(FFN / 128, TOK / 32), 256, 0, stream>>>(n2, wuT, ubf, nullptr, TOK, FFN, HID);

    // 9) SiLU * up -> bf16
    k_silu_mul<<<cdiv((long long)TOK * FFN, 256), 256, 0, stream>>>(gbf, ubf, act, (long long)TOK * FFN);

    // 10) down projection + fused residual (attn_out) -> fp32 output
    k_gemm_bf16<<<dim3(HID / 128, TOK / 32), 256, 0, stream>>>(act, wdT, out, att, TOK, HID, FFN);
}
